// Attention_72060961292740
// MI455X (gfx1250) — matrix operand order repaired and checked by reading, compiled, not run on hardware
//
#include <hip/hip_runtime.h>

typedef __bf16 bf16;
typedef __attribute__((ext_vector_type(16))) __bf16 v16bf;
typedef __attribute__((ext_vector_type(8)))  __bf16 v8bf;
typedef __attribute__((ext_vector_type(8)))  float  v8f;
typedef __attribute__((ext_vector_type(8)))  unsigned int v8u;

// ---------- helpers ----------
static __device__ __forceinline__ unsigned short bfbits(float f) {
    unsigned int u = __builtin_bit_cast(unsigned int, f);
    u += 0x7fffu + ((u >> 16) & 1u);          // round-to-nearest-even
    return (unsigned short)(u >> 16);
}
static __device__ __forceinline__ bf16 f2bf(float f) {
    unsigned short s = bfbits(f);
    return __builtin_bit_cast(bf16, s);
}
static __device__ __forceinline__ unsigned int pack2(float a, float b) {
    return (unsigned int)bfbits(a) | ((unsigned int)bfbits(b) << 16);
}
static __device__ __forceinline__ v8bf  ld8 (const bf16* p) { return *(const v8bf*)p; }
static __device__ __forceinline__ v16bf ld16(const bf16* p) { return *(const v16bf*)p; }
static __device__ __forceinline__ v16bf join16(v8bf lo, v8bf hi) {
    return __builtin_shufflevector(lo, hi, 0,1,2,3,4,5,6,7,8,9,10,11,12,13,14,15);
}
static __device__ __forceinline__ v8f wmma_bf16(v16bf a, v16bf b, v8f c) {
    return __builtin_amdgcn_wmma_f32_16x16x32_bf16(false, a, false, b, (short)0, c, false, false);
}
// LDS byte offset of a generic pointer to __shared__: ISA truncates flat addr to 32b
static __device__ __forceinline__ unsigned lds_off(const void* p) {
    return (unsigned)(unsigned long long)p;
}
// async 16B global -> LDS copy (ASYNCcnt-tracked)
static __device__ __forceinline__ void async_ld16(unsigned loff, const bf16* g) {
    unsigned long long ga = (unsigned long long)g;
    asm volatile("global_load_async_to_lds_b128 %0, %1, off"
                 :: "v"(loff), "v"(ga) : "memory");
}
static __device__ __forceinline__ void wait_async0() {
    asm volatile("s_wait_asynccnt 0x0" ::: "memory");
}

// B=4, C=256, N=4096, HEADS=4, DH=32, HID=128, QKV rows=384
#define NTOK 4096
#define LROW 40   // padded LDS row stride (elements) for bank-conflict-free b128 reads

// ---------- kernel 0a: x [b,c,n] f32 -> xT [b,n,c] bf16 (LDS tile transpose) ----------
__global__ void transpose_x(const float* __restrict__ x, bf16* __restrict__ xT) {
    __shared__ float tile[32][33];
    int blk = blockIdx.x;                 // 4 * 8 * 128 = 4096 blocks
    int b  = blk >> 10;
    int rem = blk & 1023;
    int ct = rem >> 7;                    // c tile (0..7)
    int nt = rem & 127;                   // n tile (0..127)
    int c0 = ct * 32, n0 = nt * 32;
    int col = threadIdx.x & 31, row8 = threadIdx.x >> 5;
#pragma unroll
    for (int ph = 0; ph < 4; ++ph) {
        int r = ph * 8 + row8;
        tile[r][col] = x[((size_t)b * 256 + c0 + r) * NTOK + n0 + col];
    }
    __syncthreads();
#pragma unroll
    for (int ph = 0; ph < 4; ++ph) {
        int r = ph * 8 + row8;
        xT[((size_t)b * NTOK + n0 + r) * 256 + c0 + col] = f2bf(tile[col][r]);
    }
}

// ---------- kernel 0b: weights f32 -> bf16 ----------
__global__ void cvt_w(const float* __restrict__ wq, const float* __restrict__ wo,
                      bf16* __restrict__ wqb, bf16* __restrict__ wob) {
    int i = blockIdx.x * 256 + threadIdx.x;      // 131072 total
    if (i < 98304) wqb[i] = f2bf(wq[i]);
    else           wob[i - 98304] = f2bf(wo[i - 98304]);
}

// ---------- kernel 1: QKV projection. qkv[o,n] = w_qkv[o,c] * x[c,n] ----------
__global__ void qkv_gemm(const bf16* __restrict__ xT, const bf16* __restrict__ wq,
                         bf16* __restrict__ q, bf16* __restrict__ k, bf16* __restrict__ v) {
    const int lane = threadIdx.x & 31, li = lane & 15, hh = lane >> 4;
    int tile = blockIdx.x * 8 + (threadIdx.x >> 5);   // 24576 tiles: b * (24 o) * (256 n)
    int b = tile / 6144; int rem = tile - b * 6144;
    int ot = rem >> 8, nt = rem & 255;
    int ob = ot * 16, nb = nt * 16;
    const bf16* arow = xT + ((size_t)b * NTOK + nb + li) * 256;
    const bf16* brow = wq + (size_t)(ob + li) * 256;
    v8f acc = {};
#pragma unroll
    for (int kb = 0; kb < 256; kb += 32) {
        v16bf a  = ld16(arow + kb + hh * 16);
        v16bf bb = ld16(brow + kb + hh * 16);
        acc = wmma_bf16(a, bb, acc);
    }
    int o = ob + li;
    int sec = o >> 7, hd = o & 127, h = hd >> 5, d = hd & 31;
    int bh = b * 4 + h;
    const float QS = 0.17677669529663687f * 1.4426950408889634f; // 32^-0.5 * log2(e)
#pragma unroll
    for (int r = 0; r < 8; ++r) {
        int n = nb + r + 8 * hh;
        float val = acc[r];
        if (sec == 0)      q[((size_t)bh * NTOK + n) * 32 + d] = f2bf(val * QS);
        else if (sec == 1) k[((size_t)bh * NTOK + n) * 32 + d] = f2bf(val);
        else               v[((size_t)bh * 32 + d) * NTOK + n] = f2bf(val);
    }
}

// ---------- kernel 2: fused flash attention -------------------------------------------
// 8 waves/block share one (b,h); K/V 32-key blocks staged into LDS via async DMA
// (double-buffered). Each wave: 16 queries, S^T orientation so lane owns row i,
// online softmax in registers, P^T -> B operand with packs + shfl_xor(16) only.
__global__ void flash_attn(const bf16* __restrict__ q, const bf16* __restrict__ k,
                           const bf16* __restrict__ v, bf16* __restrict__ attT) {
    __shared__ bf16 kbuf[2][32 * LROW];
    __shared__ bf16 vbuf[2][32 * LROW];
    const int tid = threadIdx.x;
    const int lane = tid & 31, li = lane & 15, hh = lane >> 4, wv = tid >> 5;
    int blk = blockIdx.x;                 // 512 = 16 bh * 32 query groups
    int bh = blk >> 5, grp = blk & 31;
    int ib = (grp * 8 + wv) * 16;
    const bf16* qbase = q + (size_t)bh * NTOK * 32;
    const bf16* kbase = k + (size_t)bh * NTOK * 32;
    const bf16* vbase = v + (size_t)bh * 32 * NTOK;

    // staging assignment: 128 threads stage K (32 rows x 4 chunks), 128 stage V
    const int srow = (tid & 127) >> 2;
    const int sch  = (tid & 3) * 8;
    const bool isV = tid >= 128;
    const bf16* gsrc = isV ? (vbase + (size_t)srow * NTOK + sch)   // v[d=srow][sch+jb..]
                           : (kbase + (size_t)srow * 32 + sch);    // k[jb+srow][sch..]
    const unsigned loff[2] = {
        lds_off(isV ? &vbuf[0][srow * LROW + sch] : &kbuf[0][srow * LROW + sch]),
        lds_off(isV ? &vbuf[1][srow * LROW + sch] : &kbuf[1][srow * LROW + sch])
    };

    v16bf qf = join16(ld8(qbase + (size_t)(ib + li) * 32 + hh * 8), ld8(qbase + (size_t)(ib + li) * 32 + hh * 8 + 16));  // B operand (N=i, K=d)
    v8f acc0 = {}, acc1 = {};            // O^T tiles: d 0..15 / 16..31, cols = i
    float m_i = -3.0e38f, l_i = 0.0f;

    // prologue: stage block 0 into buffer 0
    async_ld16(loff[0], gsrc);           // k: row srow of block 0 | v: cols 0..31
    wait_async0();
    __syncthreads();

    for (int bj = 0; bj < 128; ++bj) {
        const int buf = bj & 1;
        const int jb = bj * 32;
        // issue DMA for next key block into the other buffer (overlaps compute)
        if (bj + 1 < 128) {
            size_t goff = isV ? (size_t)(jb + 32) : (size_t)(jb + 32) * 32;
            async_ld16(loff[buf ^ 1], gsrc + goff);
        }
        // ---- S^T tiles from LDS: A = k rows (M=j, K=d=32), B = qf
        const bf16* kb0 = &kbuf[buf][li * LROW + hh * 8];
        const bf16* kb1 = &kbuf[buf][(16 + li) * LROW + hh * 8];
        v16bf ka0 = join16(ld8(kb0), ld8(kb0 + 16));
        v16bf ka1 = join16(ld8(kb1), ld8(kb1 + 16));
        v8f z = {};
        v8f s0 = wmma_bf16(ka0, qf, z);   // rows j=jb+r+8hh, col i=li
        v8f s1 = wmma_bf16(ka1, qf, z);   // rows j=jb+16+r+8hh

        // ---- online softmax for row i (this lane + partner lane via xor 16)
        float mb = s0[0];
#pragma unroll
        for (int r = 1; r < 8; ++r) mb = fmaxf(mb, s0[r]);
#pragma unroll
        for (int r = 0; r < 8; ++r) mb = fmaxf(mb, s1[r]);
        mb = fmaxf(mb, __shfl_xor(mb, 16, 32));
        float m_new = fmaxf(m_i, mb);
        float alpha = exp2f(m_i - m_new);
        m_i = m_new;

        float p0[8], p1[8], sum = 0.0f;
#pragma unroll
        for (int r = 0; r < 8; ++r) { p0[r] = exp2f(s0[r] - m_new); sum += p0[r]; }
#pragma unroll
        for (int r = 0; r < 8; ++r) { p1[r] = exp2f(s1[r] - m_new); sum += p1[r]; }
        sum += __shfl_xor(sum, 16, 32);
        l_i = l_i * alpha + sum;
#pragma unroll
        for (int r = 0; r < 8; ++r) { acc0[r] *= alpha; acc1[r] *= alpha; }

        // ---- assemble P^T as B operand (N=i, K=j 0..31): 8 packs + 8 shuffles
        unsigned int pk0[4], pk1[4];
#pragma unroll
        for (int w = 0; w < 4; ++w) {
            pk0[w] = pack2(p0[2 * w], p0[2 * w + 1]);
            pk1[w] = pack2(p1[2 * w], p1[2 * w + 1]);
        }
        v8u bw;
#pragma unroll
        for (int w = 0; w < 4; ++w) {
            unsigned int x0 = __shfl_xor(pk0[w], 16, 32);
            unsigned int x1 = __shfl_xor(pk1[w], 16, 32);
            bw[w]     = pk0[w];   // low: j0..7 own  | high: j8..15 own
            bw[w + 4] = pk1[w];   // low: j16..23 own | high: j24..31 own
        }
        v16bf bp = __builtin_bit_cast(v16bf, bw);

        // ---- PV from LDS: A = v rows (M=d, K=j=32), accumulate O^T
        const bf16* vb0 = &vbuf[buf][li * LROW + hh * 8];
        const bf16* vb1 = &vbuf[buf][(16 + li) * LROW + hh * 8];
        v16bf va0 = join16(ld8(vb0), ld8(vb0 + 16));
        v16bf va1 = join16(ld8(vb1), ld8(vb1 + 16));
        acc0 = wmma_bf16(va0, bp, acc0);
        acc1 = wmma_bf16(va1, bp, acc1);

        // my next-buffer DMA landed; barrier also separates compute from overwrite
        wait_async0();
        __syncthreads();
    }

    float inv = 1.0f / l_i;
    int b = bh >> 2, h = bh & 3;
    int n = ib + li;
#pragma unroll
    for (int r = 0; r < 8; ++r) {
        int c0 = h * 32 + r + 8 * hh;        // acc0: d = r+8hh
        int c1 = c0 + 16;                    // acc1: d = 16+r+8hh
        attT[((size_t)b * NTOK + n) * 128 + c0] = f2bf(acc0[r] * inv);
        attT[((size_t)b * NTOK + n) * 128 + c1] = f2bf(acc1[r] * inv);
    }
}

// ---------- kernel 3: out[o,n] = w_out[o,c] * att[c,n] + b_out[o] ----------
__global__ void out_gemm(const bf16* __restrict__ wo, const bf16* __restrict__ attT,
                         const float* __restrict__ bias, float* __restrict__ out) {
    const int lane = threadIdx.x & 31, li = lane & 15, hh = lane >> 4;
    int tile = blockIdx.x * 8 + (threadIdx.x >> 5);   // 16384 = 4 b * 16 ot * 256 nt
    int b = tile >> 12; int rem = tile & 4095;
    int ot = rem >> 8, nt = rem & 255;
    int ob = ot * 16, nb = nt * 16;
    const bf16* arow = wo + (size_t)(ob + li) * 128;
    const bf16* brow = attT + ((size_t)b * NTOK + nb + li) * 128;
    v8f acc = {};
#pragma unroll
    for (int kb = 0; kb < 128; kb += 32) {
        v16bf a  = ld16(arow + kb + hh * 16);
        v16bf bb = ld16(brow + kb + hh * 16);
        acc = wmma_bf16(a, bb, acc);
    }
#pragma unroll
    for (int r = 0; r < 8; ++r) {
        int o = ob + r + 8 * hh;
        out[((size_t)b * 256 + o) * NTOK + nb + li] = acc[r] + bias[o];
    }
}

// ---------- launcher ----------
extern "C" void kernel_launch(void* const* d_in, const int* in_sizes, int n_in,
                              void* d_out, int out_size, void* d_ws, size_t ws_size,
                              hipStream_t stream) {
    (void)in_sizes; (void)n_in; (void)out_size; (void)ws_size;
    const float* x     = (const float*)d_in[0];
    const float* w_qkv = (const float*)d_in[1];
    const float* w_out = (const float*)d_in[2];
    const float* b_out = (const float*)d_in[3];
    float* out = (float*)d_out;

    char* ws = (char*)d_ws;
    bf16* xT  = (bf16*)(ws);                  // 8 MB
    bf16* q   = (bf16*)(ws + 8388608);        // 4 MB
    bf16* k   = (bf16*)(ws + 12582912);       // 4 MB
    bf16* v   = (bf16*)(ws + 16777216);       // 4 MB
    bf16* att = (bf16*)(ws + 20971520);       // 4 MB
    bf16* wqb = (bf16*)(ws + 25165824);       // 192 KB
    bf16* wob = (bf16*)(ws + 25362432);       // 64 KB

    transpose_x<<<4096, 256, 0, stream>>>(x, xT);
    cvt_w<<<512, 256, 0, stream>>>(w_qkv, w_out, wqb, wob);
    qkv_gemm<<<3072, 256, 0, stream>>>(xT, wqb, q, k, v);
    flash_attn<<<512, 256, 0, stream>>>(q, k, v, att);
    out_gemm<<<2048, 256, 0, stream>>>(wob, att, b_out, out);
}
